// attention_8100308320730
// MI455X (gfx1250) — matrix operand order repaired and checked by reading, not compiled, not run on hardware
//
#include <hip/hip_runtime.h>
#include <hip/hip_bf16.h>

// MI455X / gfx1250 multi-head attention, bf16 WMMA pipeline, software-pipelined.
//  K1: QKV projection (f32 in -> bf16 out, fused 64^-0.25 scale on Q/K)
//  K2: flash attention per (b,h), online softmax, WMMA QK^T and PV
//  K3: output projection + bias (bf16 x bf16 -> f32)
// All GEMM loops: double-buffered LDS tiles, prefetch next tile's global loads
// into registers during the current tile's WMMAs, batched fragment loads so the
// 4 WMMAs per group issue back-to-back.

typedef __bf16 bf16_t;
typedef __attribute__((ext_vector_type(16))) __bf16 v16bf;
typedef __attribute__((ext_vector_type(8)))  __bf16 v8bf;
typedef __attribute__((ext_vector_type(8)))  float  v8f;

#define D_MODEL 1024
#define NHEAD   16
#define ATT_DIM 64
#define SEQ     2048
#define BATCH   2
#define PROJ_ELEMS ((size_t)BATCH * NHEAD * SEQ * ATT_DIM)   // 4,194,304 bf16 per matrix

static __device__ __forceinline__ v8f zero8() {
  v8f z = {0.f, 0.f, 0.f, 0.f, 0.f, 0.f, 0.f, 0.f};
  return z;
}

static __device__ __forceinline__ v8f wmma_bf16f32(v16bf a, v16bf b, v8f c) {
  return __builtin_amdgcn_wmma_f32_16x16x32_bf16(false, a, false, b, (short)0, c,
                                                 false, false);
}

// Assemble a 16-element bf16 fragment from two contiguous 8-element (16B) chunks.
static __device__ __forceinline__ v16bf frag_from(const bf16_t* lo, const bf16_t* hi) {
  union { v16bf v; v8bf h[2]; } u;
  u.h[0] = *reinterpret_cast<const v8bf*>(lo);
  u.h[1] = *reinterpret_cast<const v8bf*>(hi);
  return u.v;
}

// A-fragment (16x32, row M = lane&15) straight from an f32 row in global memory.
// Element i -> K = 16g+i.
static __device__ __forceinline__ v16bf afrag_from_f32(const float* rowp, int g) {
  const float4* p0 = reinterpret_cast<const float4*>(rowp + 16 * g);
  const float4* p1 = reinterpret_cast<const float4*>(rowp + 16 * g + 8);
  float4 a0 = p0[0], a1 = p0[1], b0 = p1[0], b1 = p1[1];
  v16bf f;
  f[0]  = (bf16_t)a0.x; f[1]  = (bf16_t)a0.y; f[2]  = (bf16_t)a0.z; f[3]  = (bf16_t)a0.w;
  f[4]  = (bf16_t)a1.x; f[5]  = (bf16_t)a1.y; f[6]  = (bf16_t)a1.z; f[7]  = (bf16_t)a1.w;
  f[8]  = (bf16_t)b0.x; f[9]  = (bf16_t)b0.y; f[10] = (bf16_t)b0.z; f[11] = (bf16_t)b0.w;
  f[12] = (bf16_t)b1.x; f[13] = (bf16_t)b1.y; f[14] = (bf16_t)b1.z; f[15] = (bf16_t)b1.w;
  return f;
}

static __device__ __forceinline__ v8bf cvt8(float4 c0, float4 c1) {
  v8bf d;
  d[0] = (bf16_t)c0.x; d[1] = (bf16_t)c0.y; d[2] = (bf16_t)c0.z; d[3] = (bf16_t)c0.w;
  d[4] = (bf16_t)c1.x; d[5] = (bf16_t)c1.y; d[6] = (bf16_t)c1.z; d[7] = (bf16_t)c1.w;
  return d;
}

// ---------------------------------------------------------------------------
// Kernel 1: P = X @ W^T, scaled, converted to bf16, stored [b][h][s][e].
// Grid: (16 n-tiles of 64, 32 m-tiles of 128, 3 matrices). Block: 256 (8 waves).
// ---------------------------------------------------------------------------
__global__ __launch_bounds__(256) void qkv_proj_kernel(
    const float* __restrict__ q, const float* __restrict__ k, const float* __restrict__ v,
    const float* __restrict__ Wq, const float* __restrict__ Wk, const float* __restrict__ Wv,
    bf16_t* __restrict__ Qp, bf16_t* __restrict__ Kp, bf16_t* __restrict__ Vp) {
  __shared__ bf16_t Bt[2][64 * 40];  // W tile [N=64][K=32], stride 40, ping-pong

  const int z = blockIdx.z;
  const float* X = (z == 0) ? q : (z == 1) ? k : v;
  const float* W = (z == 0) ? Wq : (z == 1) ? Wk : Wv;
  bf16_t* P = (z == 0) ? Qp : (z == 1) ? Kp : Vp;
  const float scale = (z == 2) ? 1.0f : 0.35355339059327373f;  // 64^-0.25 on Q,K

  const int tid = threadIdx.x;
  const int w = tid >> 5, lane = tid & 31;
  const int g = lane >> 4, mr = lane & 15, nn = lane & 15;
  const int mbase = blockIdx.y * 128 + w * 16;
  const int nbase = blockIdx.x * 64;
  const int n_ld = tid >> 2;          // 0..63: W tile row this thread stages
  const int kk_ld = (tid & 3) * 8;    // 0/8/16/24: K chunk

  const float* Wrow = W + (size_t)(nbase + n_ld) * D_MODEL + kk_ld;
  const float* Xrow = X + (size_t)(mbase + mr) * D_MODEL;

  v8f acc[4];
#pragma unroll
  for (int i = 0; i < 4; i++) acc[i] = zero8();

  // Prologue: stage tile 0
  {
    const float4* src = reinterpret_cast<const float4*>(Wrow);
    *reinterpret_cast<v8bf*>(&Bt[0][n_ld * 40 + kk_ld]) = cvt8(src[0], src[1]);
  }
  v16bf af = afrag_from_f32(Xrow, g);
  __syncthreads();

  for (int it = 0; it < 32; ++it) {
    const int last = (it == 31);
    v8bf wnext;
    v16bf afnext;
    if (!last) {  // issue next tile's global loads before touching this tile
      const float4* src = reinterpret_cast<const float4*>(Wrow + (it + 1) * 32);
      wnext = cvt8(src[0], src[1]);
      afnext = afrag_from_f32(Xrow + (it + 1) * 32, g);
    }
    const bf16_t* buf = Bt[it & 1];
    v16bf b0 = frag_from(&buf[(0  + nn) * 40 + 16 * g], &buf[(0  + nn) * 40 + 16 * g + 8]);
    v16bf b1 = frag_from(&buf[(16 + nn) * 40 + 16 * g], &buf[(16 + nn) * 40 + 16 * g + 8]);
    v16bf b2 = frag_from(&buf[(32 + nn) * 40 + 16 * g], &buf[(32 + nn) * 40 + 16 * g + 8]);
    v16bf b3 = frag_from(&buf[(48 + nn) * 40 + 16 * g], &buf[(48 + nn) * 40 + 16 * g + 8]);
    acc[0] = wmma_bf16f32(af, b0, acc[0]);
    acc[1] = wmma_bf16f32(af, b1, acc[1]);
    acc[2] = wmma_bf16f32(af, b2, acc[2]);
    acc[3] = wmma_bf16f32(af, b3, acc[3]);
    if (!last) {
      *reinterpret_cast<v8bf*>(&Bt[(it + 1) & 1][n_ld * 40 + kk_ld]) = wnext;
      af = afnext;
      __syncthreads();
    }
  }

#pragma unroll
  for (int nt = 0; nt < 4; nt++) {
#pragma unroll
    for (int r = 0; r < 8; r++) {
      int row = mbase + r + 8 * g;        // global row in [0, b*s)
      int col = nbase + nt * 16 + nn;     // output feature = h*64 + e
      int b = row >> 11, s = row & 2047;
      int h = col >> 6, e = col & 63;
      P[(((size_t)(b * NHEAD + h)) * SEQ + s) * ATT_DIM + e] =
          (bf16_t)(acc[nt][r] * scale);
    }
  }
}

// ---------------------------------------------------------------------------
// Kernel 2: flash attention. Grid: (16 q-tiles of 128, 32 bh). Block: 256.
// Each wave owns 16 q rows; waves share double-buffered LDS K/V tiles (32 keys).
// ---------------------------------------------------------------------------
__global__ __launch_bounds__(256) void flash_attn_kernel(
    const bf16_t* __restrict__ Qp, const bf16_t* __restrict__ Kp,
    const bf16_t* __restrict__ Vp, bf16_t* __restrict__ Op) {
  __shared__ bf16_t Ks[2][32 * 72];    // [k_local][e], stride 72
  __shared__ bf16_t Vt[2][64 * 40];    // [e][k_local], stride 40 (transposed)
  __shared__ bf16_t Pst[8 * 16 * 40];  // per-wave P staging [16][32], stride 40

  const int tid = threadIdx.x;
  const int w = tid >> 5, lane = tid & 31;
  const int g = lane >> 4, mr = lane & 15, nn = lane & 15;
  const int bh = blockIdx.y;
  const size_t base = (size_t)bh * SEQ * ATT_DIM;
  const bf16_t* Qb = Qp + base;
  const bf16_t* Kb = Kp + base;
  const bf16_t* Vb = Vp + base;
  const int qbase = blockIdx.x * 128 + w * 16;
  const int rr = tid >> 3;         // 0..31: staged key row
  const int cc = (tid & 7) * 8;    // 0..56: e chunk

  // Q A-fragments for e=0..31 and e=32..63 (loaded once, reused 64x)
  const bf16_t* qrow = Qb + (size_t)(qbase + mr) * ATT_DIM;
  v16bf qf0 = frag_from(qrow + 16 * g, qrow + 16 * g + 8);
  v16bf qf1 = frag_from(qrow + 32 + 16 * g, qrow + 32 + 16 * g + 8);

  v8f o0 = zero8(), o1 = zero8(), o2 = zero8(), o3 = zero8();
  float mrow[8], lrow[8];
#pragma unroll
  for (int r = 0; r < 8; r++) { mrow[r] = -1e30f; lrow[r] = 0.f; }

  bf16_t* pw = &Pst[w * 16 * 40];
  const bf16_t* Krow = Kb + (size_t)rr * ATT_DIM + cc;
  const bf16_t* Vrow = Vb + (size_t)rr * ATT_DIM + cc;

  // Prologue: stage key-tile 0
  {
    v8bf kreg = *reinterpret_cast<const v8bf*>(Krow);
    v8bf vreg = *reinterpret_cast<const v8bf*>(Vrow);
    *reinterpret_cast<v8bf*>(&Ks[0][rr * 72 + cc]) = kreg;
#pragma unroll
    for (int j = 0; j < 8; j++) Vt[0][(cc + j) * 40 + rr] = vreg[j];
  }
  __syncthreads();

  for (int it = 0; it < SEQ / 32; ++it) {
    const int last = (it == SEQ / 32 - 1);
    v8bf knext, vnext;
    if (!last) {  // prefetch next 32-key tile while computing this one
      knext = *reinterpret_cast<const v8bf*>(Krow + (size_t)(it + 1) * 32 * ATT_DIM);
      vnext = *reinterpret_cast<const v8bf*>(Vrow + (size_t)(it + 1) * 32 * ATT_DIM);
    }
    const bf16_t* ks = Ks[it & 1];
    const bf16_t* vt = Vt[it & 1];

    // S = Q (16x64) * K_tile^T (64x32): batched frags, then 4 WMMA
    v16bf kb0 = frag_from(&ks[nn * 72 + 16 * g],      &ks[nn * 72 + 16 * g + 8]);
    v16bf kb1 = frag_from(&ks[nn * 72 + 32 + 16 * g], &ks[nn * 72 + 32 + 16 * g + 8]);
    v16bf kc0 = frag_from(&ks[(16 + nn) * 72 + 16 * g],      &ks[(16 + nn) * 72 + 16 * g + 8]);
    v16bf kc1 = frag_from(&ks[(16 + nn) * 72 + 32 + 16 * g], &ks[(16 + nn) * 72 + 32 + 16 * g + 8]);
    v8f s0 = zero8(), s1 = zero8();
    s0 = wmma_bf16f32(qf0, kb0, s0);
    s1 = wmma_bf16f32(qf0, kc0, s1);
    s0 = wmma_bf16f32(qf1, kb1, s0);
    s1 = wmma_bf16f32(qf1, kc1, s1);

    // Online softmax. C-layout: VGPR r <-> row r+8g, lane%16 <-> key.
    float al[8];
#pragma unroll
    for (int r = 0; r < 8; r++) {
      float t = fmaxf(s0[r], s1[r]);
      t = fmaxf(t, __shfl_xor(t, 1, 32));
      t = fmaxf(t, __shfl_xor(t, 2, 32));
      t = fmaxf(t, __shfl_xor(t, 4, 32));
      t = fmaxf(t, __shfl_xor(t, 8, 32));
      float mn = fmaxf(mrow[r], t);
      float a = __expf(mrow[r] - mn);
      float p0 = __expf(s0[r] - mn);
      float p1 = __expf(s1[r] - mn);
      s0[r] = p0;
      s1[r] = p1;
      float rs = p0 + p1;
      rs += __shfl_xor(rs, 1, 32);
      rs += __shfl_xor(rs, 2, 32);
      rs += __shfl_xor(rs, 4, 32);
      rs += __shfl_xor(rs, 8, 32);
      lrow[r] = lrow[r] * a + rs;
      mrow[r] = mn;
      al[r] = a;
    }
#pragma unroll
    for (int r = 0; r < 8; r++) {
      o0[r] *= al[r]; o1[r] *= al[r]; o2[r] *= al[r]; o3[r] *= al[r];
    }

    // Route P through per-wave LDS: C-layout -> A-fragment layout (16x32 bf16).
    // Wave-private slice; LDS ops are in-order per wave, no barrier needed.
#pragma unroll
    for (int r = 0; r < 8; r++) {
      pw[(r + 8 * g) * 40 + nn] = (bf16_t)s0[r];
      pw[(r + 8 * g) * 40 + 16 + nn] = (bf16_t)s1[r];
    }
    v16bf pf = frag_from(&pw[mr * 40 + 16 * g], &pw[mr * 40 + 16 * g + 8]);

    // O += P (16x32) * V_tile (32x64): batched frags, then 4 WMMA
    v16bf v0 = frag_from(&vt[(0  + nn) * 40 + 16 * g], &vt[(0  + nn) * 40 + 16 * g + 8]);
    v16bf v1 = frag_from(&vt[(16 + nn) * 40 + 16 * g], &vt[(16 + nn) * 40 + 16 * g + 8]);
    v16bf v2 = frag_from(&vt[(32 + nn) * 40 + 16 * g], &vt[(32 + nn) * 40 + 16 * g + 8]);
    v16bf v3 = frag_from(&vt[(48 + nn) * 40 + 16 * g], &vt[(48 + nn) * 40 + 16 * g + 8]);
    o0 = wmma_bf16f32(pf, v0, o0);
    o1 = wmma_bf16f32(pf, v1, o1);
    o2 = wmma_bf16f32(pf, v2, o2);
    o3 = wmma_bf16f32(pf, v3, o3);

    if (!last) {  // stage next tile into the other buffer; one barrier per iter
      *reinterpret_cast<v8bf*>(&Ks[(it + 1) & 1][rr * 72 + cc]) = knext;
#pragma unroll
      for (int j = 0; j < 8; j++) Vt[(it + 1) & 1][(cc + j) * 40 + rr] = vnext[j];
      __syncthreads();
    }
  }

  // Normalize and write attention output [b][s][h*64+e] as bf16
  float inv[8];
#pragma unroll
  for (int r = 0; r < 8; r++) inv[r] = 1.0f / lrow[r];
  const int b = bh >> 4, h = bh & 15;
  bf16_t* Ob = Op + (size_t)b * SEQ * D_MODEL;
#pragma unroll
  for (int r = 0; r < 8; r++) {
    int srow = qbase + r + 8 * g;
    size_t rb = (size_t)srow * D_MODEL + h * 64 + nn;
    Ob[rb + 0]  = (bf16_t)(o0[r] * inv[r]);
    Ob[rb + 16] = (bf16_t)(o1[r] * inv[r]);
    Ob[rb + 32] = (bf16_t)(o2[r] * inv[r]);
    Ob[rb + 48] = (bf16_t)(o3[r] * inv[r]);
  }
}

// ---------------------------------------------------------------------------
// Kernel 3: out = AttO @ Wo^T + bo, f32 result. Grid: (16, 32). Block: 256.
// ---------------------------------------------------------------------------
__global__ __launch_bounds__(256) void out_proj_kernel(
    const bf16_t* __restrict__ A, const float* __restrict__ Wo,
    const float* __restrict__ bo, float* __restrict__ out) {
  __shared__ bf16_t Bt[2][64 * 40];

  const int tid = threadIdx.x;
  const int w = tid >> 5, lane = tid & 31;
  const int g = lane >> 4, mr = lane & 15, nn = lane & 15;
  const int mbase = blockIdx.y * 128 + w * 16;
  const int nbase = blockIdx.x * 64;
  const int n_ld = tid >> 2;
  const int kk_ld = (tid & 3) * 8;

  const float* Wrow = Wo + (size_t)(nbase + n_ld) * D_MODEL + kk_ld;
  const bf16_t* Arow = A + (size_t)(mbase + mr) * D_MODEL;

  v8f acc[4];
#pragma unroll
  for (int i = 0; i < 4; i++) acc[i] = zero8();

  {
    const float4* src = reinterpret_cast<const float4*>(Wrow);
    *reinterpret_cast<v8bf*>(&Bt[0][n_ld * 40 + kk_ld]) = cvt8(src[0], src[1]);
  }
  v16bf af = frag_from(Arow + 16 * g, Arow + 16 * g + 8);
  __syncthreads();

  for (int it = 0; it < 32; ++it) {
    const int last = (it == 31);
    v8bf wnext;
    v16bf afnext;
    if (!last) {
      const float4* src = reinterpret_cast<const float4*>(Wrow + (it + 1) * 32);
      wnext = cvt8(src[0], src[1]);
      const bf16_t* ar = Arow + (it + 1) * 32;
      afnext = frag_from(ar + 16 * g, ar + 16 * g + 8);
    }
    const bf16_t* buf = Bt[it & 1];
    v16bf b0 = frag_from(&buf[(0  + nn) * 40 + 16 * g], &buf[(0  + nn) * 40 + 16 * g + 8]);
    v16bf b1 = frag_from(&buf[(16 + nn) * 40 + 16 * g], &buf[(16 + nn) * 40 + 16 * g + 8]);
    v16bf b2 = frag_from(&buf[(32 + nn) * 40 + 16 * g], &buf[(32 + nn) * 40 + 16 * g + 8]);
    v16bf b3 = frag_from(&buf[(48 + nn) * 40 + 16 * g], &buf[(48 + nn) * 40 + 16 * g + 8]);
    acc[0] = wmma_bf16f32(af, b0, acc[0]);
    acc[1] = wmma_bf16f32(af, b1, acc[1]);
    acc[2] = wmma_bf16f32(af, b2, acc[2]);
    acc[3] = wmma_bf16f32(af, b3, acc[3]);
    if (!last) {
      *reinterpret_cast<v8bf*>(&Bt[(it + 1) & 1][n_ld * 40 + kk_ld]) = wnext;
      af = afnext;
      __syncthreads();
    }
  }

#pragma unroll
  for (int nt = 0; nt < 4; nt++) {
#pragma unroll
    for (int r = 0; r < 8; r++) {
      int row = mbase + r + 8 * g;
      int col = nbase + nt * 16 + nn;
      out[(size_t)row * D_MODEL + col] = acc[nt][r] + bo[col];
    }
  }
}

// ---------------------------------------------------------------------------
extern "C" void kernel_launch(void* const* d_in, const int* in_sizes, int n_in,
                              void* d_out, int out_size, void* d_ws, size_t ws_size,
                              hipStream_t stream) {
  (void)in_sizes; (void)n_in; (void)out_size; (void)ws_size;
  const float* q  = (const float*)d_in[0];
  const float* k  = (const float*)d_in[1];
  const float* v  = (const float*)d_in[2];
  const float* Wq = (const float*)d_in[3];
  const float* Wk = (const float*)d_in[4];
  const float* Wv = (const float*)d_in[5];
  const float* Wo = (const float*)d_in[6];
  const float* bo = (const float*)d_in[7];

  // Workspace layout (bf16): Qp | Kp | Vp | AttO  -> 4 * 8 MB = 32 MB
  bf16_t* ws = (bf16_t*)d_ws;
  bf16_t* Qp = ws;
  bf16_t* Kp = ws + PROJ_ELEMS;
  bf16_t* Vp = ws + 2 * PROJ_ELEMS;
  bf16_t* Ao = ws + 3 * PROJ_ELEMS;

  qkv_proj_kernel<<<dim3(16, 32, 3), 256, 0, stream>>>(q, k, v, Wq, Wk, Wv, Qp, Kp, Vp);
  flash_attn_kernel<<<dim3(16, 32), 256, 0, stream>>>(Qp, Kp, Vp, Ao);
  out_proj_kernel<<<dim3(16, 32), 256, 0, stream>>>(Ao, Wo, bo, (float*)d_out);
}
